// GPTOSSRouter_18580028523158
// MI455X (gfx1250) — matrix operand order repaired and checked by reading, not compiled, not run on hardware
//
#include <hip/hip_runtime.h>
#include <hip/hip_bf16.h>
#include <math.h>

// ---------------- problem constants (from reference) ----------------
#define NUM_EXPERTS 128
#define TOP_K       4
#define HIDDEN      2880
#define NTOKENS     8192

#define TTILE   32          // tokens per block
#define LSTRIDE 129         // LDS row stride (floats) to avoid bank conflicts

typedef __attribute__((ext_vector_type(16))) __bf16 v16bf;
typedef __attribute__((ext_vector_type(8)))  float  v8f;

// ---------------- weight fp32 -> bf16 hi/lo split -------------------
__global__ __launch_bounds__(256) void cvt_weight_kernel(
    const float* __restrict__ W, __bf16* __restrict__ whi,
    __bf16* __restrict__ wlo, int n) {
  int i = blockIdx.x * blockDim.x + threadIdx.x;
  if (i < n) {
    float w  = W[i];
    __bf16 h = (__bf16)w;               // RNE high part
    whi[i]   = h;
    wlo[i]   = (__bf16)(w - (float)h);  // residual
  }
}

__device__ __forceinline__ v8f wmma_bf16(v16bf a, v16bf b, v8f c) {
  return __builtin_amdgcn_wmma_f32_16x16x32_bf16(
      /*neg_a=*/false, a, /*neg_b=*/false, b,
      /*c_mod=*/(short)0, c, /*reuse_a=*/false, /*reuse_b=*/false);
}

// ---------------- main router kernel --------------------------------
// block = 256 threads = 8 waves; block computes 32 tokens x 128 experts.
// wave w: M-tile = w&1 (16 tokens), N-tiles = (w>>1)*2 .. +1 (32 experts).
__global__ __launch_bounds__(256) void router_kernel(
    const float*  __restrict__ X,     // [NTOKENS, HIDDEN] fp32
    const __bf16* __restrict__ Whi,   // [E, HIDDEN] bf16 hi
    const __bf16* __restrict__ Wlo,   // [E, HIDDEN] bf16 lo
    const float*  __restrict__ bias,  // [E]
    float* __restrict__ scores,       // [NTOKENS, E]
    float* __restrict__ idxout) {     // [NTOKENS, TOP_K] (as float)
  __shared__ float lds[TTILE * LSTRIDE];

  const int tid  = threadIdx.x;
  const int wave = tid >> 5;
  const int lane = tid & 31;
  const int lrow = lane & 15;   // row-in-tile (A) / col-in-tile (B)
  const int half = lane >> 4;   // K-half selector per ISA fragment layout
  const int mt   = wave & 1;            // M tile within block (0..1)
  const int nb   = (wave >> 1) * 2;     // first N tile (0,2,4,6)

  const int tokrow = blockIdx.x * TTILE + mt * 16 + lrow;
  const float* arow = X + (size_t)tokrow * HIDDEN;
  const int c0 = half * 16;       // A chunk0: K = c0..c0+7
  const int c1 = half * 16 + 8;   // A chunk1: K = c1..c1+7

  const size_t boff0 = (size_t)((nb + 0) * 16 + lrow) * HIDDEN + half * 16;
  const size_t boff1 = (size_t)((nb + 1) * 16 + lrow) * HIDDEN + half * 16;

  v8f acc0 = {};
  v8f acc1 = {};

  for (int k0 = 0; k0 < HIDDEN; k0 += 32) {
    // prefetch next K-tile of activations (global_prefetch_b8)
    if (k0 + 32 < HIDDEN) __builtin_prefetch(arow + k0 + 32 + c0, 0, 3);

    // ---- A fragment: 16 fp32 -> bf16 hi/lo (contiguous K = 16*half + i, same order as B) ----
    float4 t0 = *(const float4*)(arow + k0 + c0);
    float4 t1 = *(const float4*)(arow + k0 + c0 + 4);
    float4 t2 = *(const float4*)(arow + k0 + c1);
    float4 t3 = *(const float4*)(arow + k0 + c1 + 4);
    float av[16] = {t0.x, t0.y, t0.z, t0.w, t1.x, t1.y, t1.z, t1.w,
                    t2.x, t2.y, t2.z, t2.w, t3.x, t3.y, t3.z, t3.w};
    v16bf ahi, alo;
#pragma unroll
    for (int i = 0; i < 16; ++i) {
      __bf16 h = (__bf16)av[i];
      ahi[i] = h;
      alo[i] = (__bf16)(av[i] - (float)h);
    }

    // ---- B fragments: contiguous 32B bf16 loads from pre-split weights ----
    v16bf bh0 = *(const v16bf*)(Whi + boff0 + k0);
    v16bf bl0 = *(const v16bf*)(Wlo + boff0 + k0);
    v16bf bh1 = *(const v16bf*)(Whi + boff1 + k0);
    v16bf bl1 = *(const v16bf*)(Wlo + boff1 + k0);

    // ---- 3-term split accumulation: ~fp32 accuracy on bf16 pipes ----
    acc0 = wmma_bf16(ahi, bh0, acc0);
    acc0 = wmma_bf16(alo, bh0, acc0);
    acc0 = wmma_bf16(ahi, bl0, acc0);
    acc1 = wmma_bf16(ahi, bh1, acc1);
    acc1 = wmma_bf16(alo, bh1, acc1);
    acc1 = wmma_bf16(ahi, bl1, acc1);
  }

  // ---- C layout: VGPR j -> M = j + 8*half, N = lane-col. Park in LDS. ----
#pragma unroll
  for (int j = 0; j < 8; ++j) {
    int m = mt * 16 + half * 8 + j;
    lds[m * LSTRIDE + (nb + 0) * 16 + lrow] = acc0[j];
    lds[m * LSTRIDE + (nb + 1) * 16 + lrow] = acc1[j];
  }
  __syncthreads();

  // ---- phase 2: per-token bias add, top-4, softmax, scatter ----
  if (tid < TTILE) {
    const float* row = lds + tid * LSTRIDE;
    const size_t g = (size_t)blockIdx.x * TTILE + tid;

    float v0 = -3.4e38f, v1 = -3.4e38f, v2 = -3.4e38f, v3 = -3.4e38f;
    int   i0 = 0, i1 = 0, i2 = 0, i3 = 0;
    for (int e = 0; e < NUM_EXPERTS; ++e) {
      float v = row[e] + bias[e];
      if (v > v3) {
        if (v > v2) {
          if (v > v1) {
            if (v > v0) { v3=v2;i3=i2; v2=v1;i2=i1; v1=v0;i1=i0; v0=v;i0=e; }
            else        { v3=v2;i3=i2; v2=v1;i2=i1; v1=v; i1=e; }
          } else        { v3=v2;i3=i2; v2=v; i2=e; }
        } else          { v3=v; i3=e; }
      }
    }
    // softmax over the sorted top-4 (v0 is the max)
    float e0 = expf(v0 - v0), e1 = expf(v1 - v0),
          e2 = expf(v2 - v0), e3 = expf(v3 - v0);
    float inv = 1.0f / (e0 + e1 + e2 + e3);
    float p0 = e0 * inv, p1 = e1 * inv, p2 = e2 * inv, p3 = e3 * inv;

    float* orow = scores + g * NUM_EXPERTS;
#pragma unroll 4
    for (int e = 0; e < NUM_EXPERTS; e += 4) {
      float4 o;
      o.x = (e+0==i0)?p0:(e+0==i1)?p1:(e+0==i2)?p2:(e+0==i3)?p3:0.0f;
      o.y = (e+1==i0)?p0:(e+1==i1)?p1:(e+1==i2)?p2:(e+1==i3)?p3:0.0f;
      o.z = (e+2==i0)?p0:(e+2==i1)?p1:(e+2==i2)?p2:(e+2==i3)?p3:0.0f;
      o.w = (e+3==i0)?p0:(e+3==i1)?p1:(e+3==i2)?p2:(e+3==i3)?p3:0.0f;
      *(float4*)(orow + e) = o;
    }
    idxout[g * TOP_K + 0] = (float)i0;
    idxout[g * TOP_K + 1] = (float)i1;
    idxout[g * TOP_K + 2] = (float)i2;
    idxout[g * TOP_K + 3] = (float)i3;
  }
}

// ---------------- launch --------------------------------------------
extern "C" void kernel_launch(void* const* d_in, const int* in_sizes, int n_in,
                              void* d_out, int out_size, void* d_ws, size_t ws_size,
                              hipStream_t stream) {
  const float* X    = (const float*)d_in[0];  // hidden_states [8192,2880]
  const float* W    = (const float*)d_in[1];  // weight [128,2880]
  const float* bias = (const float*)d_in[2];  // bias [128]

  float* scores = (float*)d_out;                        // [8192,128]
  float* idxout = scores + (size_t)NTOKENS * NUM_EXPERTS; // [8192,4] as float

  __bf16* whi = (__bf16*)d_ws;
  __bf16* wlo = whi + (size_t)NUM_EXPERTS * HIDDEN;     // 1.47 MB total in ws

  const int nw = NUM_EXPERTS * HIDDEN;
  cvt_weight_kernel<<<(nw + 255) / 256, 256, 0, stream>>>(W, whi, wlo, nw);

  router_kernel<<<NTOKENS / TTILE, 256, 0, stream>>>(X, whi, wlo, bias,
                                                     scores, idxout);
}
